// CausalSelfAttentionMuP_69621419868556
// MI455X (gfx1250) — matrix operand order repaired and checked by reading, not compiled, not run on hardware
//
#include <hip/hip_runtime.h>
#include <hip/hip_bf16.h>
#include <stdint.h>

typedef __bf16 bf16;
typedef __attribute__((ext_vector_type(16))) __bf16 v16bf;
typedef __attribute__((ext_vector_type(8)))  __bf16 bf16x8;
typedef __attribute__((ext_vector_type(8)))  float  f32x8;
typedef __attribute__((ext_vector_type(8)))  float  v8f;
typedef __attribute__((ext_vector_type(4)))  unsigned int u32x4;
typedef __attribute__((ext_vector_type(8)))  int i32x8;
typedef __attribute__((ext_vector_type(4)))  int i32x4;

#define T_SEQ 2048
#define C_DIM 1024
#define NHEAD 16
#define DHEAD 64

// ---------- 8-element loads with optional on-the-fly bf16 conversion ----------
__device__ __forceinline__ bf16x8 ld8cvt(const bf16* p) { return *(const bf16x8*)p; }
__device__ __forceinline__ bf16x8 ld8cvt(const float* p) {
  f32x8 f = *(const f32x8*)p;
  bf16x8 r;
#pragma unroll
  for (int i = 0; i < 8; ++i) r[i] = (bf16)f[i];
  return r;
}

// A-fragment (16x32 bf16, MxK): lane l row M=l%16, half=l/16.
// element i -> K = 16*(i/8) + 8*half + (i%8)   (ISA 7.12.2)
template <typename T>
__device__ __forceinline__ v16bf load_frag_a(const T* base, int ld, int row, int k0, int half) {
  const T* p = base + (size_t)row * ld + k0 + 8 * half;
  bf16x8 lo = ld8cvt(p);
  bf16x8 hi = ld8cvt(p + 16);
  v16bf a;
#pragma unroll
  for (int i = 0; i < 8; ++i) { a[i] = lo[i]; a[i + 8] = hi[i]; }
  return a;
}

// B-fragment (32x16 bf16, KxN): lane l col N=l%16, half=l/16; element i -> K = 16*(i/8) + 8*half + (i%8)
template <typename T>
__device__ __forceinline__ v16bf load_frag_b(const T* base, int ld, int row, int k0, int half) {
  const T* p = base + (size_t)row * ld + k0 + 8 * half;
  bf16x8 lo = ld8cvt(p);
  bf16x8 hi = ld8cvt(p + 16);
  v16bf b;
#pragma unroll
  for (int i = 0; i < 8; ++i) { b[i] = lo[i]; b[i + 8] = hi[i]; }
  return b;
}

__device__ __forceinline__ v8f wmma_bf16(v16bf a, v16bf b, v8f c) {
  return __builtin_amdgcn_wmma_f32_16x16x32_bf16(false, a, false, b, (short)0, c, false, false);
}

__device__ __forceinline__ void store_out(float* p, float v) { *p = v; }
__device__ __forceinline__ void store_out(bf16* p, float v) { *p = (bf16)v; }

// ---------- LDS byte offset of a __shared__ object (AS3 ptrtoint) ----------
__device__ __forceinline__ unsigned lds_offset(const void* p) {
  return (unsigned)(unsigned long long)(__attribute__((address_space(3))) const void*)p;
}

// ---------- TDM: 2D tile (tile_x elems wide, tile_y rows) global -> LDS ----------
// D# per CDNA5 ISA: group0 {count=1, lds_addr, global_addr, type=2},
// group1 {data_size=1(2B), tensor_dim0=tile_x, tensor_dim1=tile_y, tile dims, dim0 stride}
__device__ __forceinline__ void tdm_load_2d(unsigned lds_byte_off, const bf16* gptr,
                                            unsigned tile_x, unsigned tile_y,
                                            unsigned long long row_stride_elems) {
  unsigned long long ga = (unsigned long long)(size_t)gptr;
  unsigned long long stride = row_stride_elems;  // in data_size units
  u32x4 g0;
  g0[0] = 1u;                                            // count=1 (valid), user mode
  g0[1] = lds_byte_off;                                  // lds_addr
  g0[2] = (unsigned)ga;                                  // global_addr[31:0]
  g0[3] = (unsigned)((ga >> 32) & 0x1FFFFFFull) | (2u << 30);  // addr[56:32] | type=2
  i32x8 g1;
  g1[0] = (int)(1u << 16);                               // data_size=1 -> 2 bytes
  g1[1] = (int)((tile_x & 0xFFFFu) << 16);               // tensor_dim0[15:0]
  g1[2] = (int)((tile_x >> 16) | ((tile_y & 0xFFFFu) << 16));  // dim0 hi | dim1 lo
  g1[3] = (int)((tile_y >> 16) | (tile_x << 16));        // dim1 hi | tile_dim0
  g1[4] = (int)(tile_y & 0xFFFFu);                       // tile_dim1 (tile_dim2=0)
  g1[5] = (int)(stride & 0xFFFFFFFFull);                 // tensor_dim0_stride lo
  g1[6] = (int)((stride >> 32) & 0xFFFFull);             // stride hi | dim1_stride lo=0
  g1[7] = 0;
  i32x4 z4 = {0, 0, 0, 0};
#if defined(__clang_major__) && (__clang_major__ >= 23)
  i32x8 z8 = {0, 0, 0, 0, 0, 0, 0, 0};
  __builtin_amdgcn_tensor_load_to_lds(g0, g1, z4, z4, z8, 0);
#else
  __builtin_amdgcn_tensor_load_to_lds(g0, g1, z4, z4, 0);
#endif
}

// ---------------------------------------------------------------------
// fp32 -> bf16 bulk conversion (one-shot; removes cvt VALU from GEMM loop)
// ---------------------------------------------------------------------
__global__ __launch_bounds__(256) void cvt_bf16_kernel(const float* __restrict__ src,
                                                       bf16* __restrict__ dst, int n8) {
  int i = blockIdx.x * blockDim.x + threadIdx.x;
  if (i < n8) {
    f32x8 f = *(const f32x8*)(src + (size_t)i * 8);
    bf16x8 r;
#pragma unroll
    for (int j = 0; j < 8; ++j) r[j] = (bf16)f[j];
    *(bf16x8*)(dst + (size_t)i * 8) = r;
  }
}

// ---------------------------------------------------------------------
// TDM-fed GEMM: out[m][n] = out_scale * sum_k A[m][k]*W[n][k]
// A/W bf16 row-major [*,K]. Block 256thr/8 waves -> 128x128 tile; wave 64x32.
// Double-buffered LDS tiles filled by tensor_load_to_lds (wave 0), TENSORcnt
// + barrier handoff; fragments via ds_read_b128.
// ---------------------------------------------------------------------
template <typename TO>
__global__ __launch_bounds__(256) void gemm_tdm_kernel(
    const bf16* __restrict__ A, const bf16* __restrict__ W, TO* __restrict__ out,
    int K, float out_scale, int64_t Sb, int64_t St, int64_t Sh, int64_t Sd) {
  __shared__ bf16 smem[2 * 8192];  // [buf][A 4096 | W 4096] = 32 KB

  const int lane = threadIdx.x & 31;
  const int w    = threadIdx.x >> 5;
  const int half = lane >> 4;
  const int lr   = lane & 15;
  const int m_blk = blockIdx.y * 128;
  const int n_blk = blockIdx.x * 128;
  const int mw = (w & 1) * 64;
  const int nw = (w >> 1) * 32;

  const unsigned lds0 = lds_offset(smem);

  v8f acc[4][2];
#pragma unroll
  for (int mi = 0; mi < 4; ++mi)
#pragma unroll
    for (int ni = 0; ni < 2; ++ni)
#pragma unroll
      for (int r = 0; r < 8; ++r) acc[mi][ni][r] = 0.0f;

  if (w == 0) {  // prologue: DMA first K tile
    tdm_load_2d(lds0,        A + (size_t)m_blk * K, 32, 128, (unsigned long long)K);
    tdm_load_2d(lds0 + 8192, W + (size_t)n_blk * K, 32, 128, (unsigned long long)K);
  }

  int cur = 0;
  for (int k0 = 0; k0 < K; k0 += 32) {
    if (w == 0) __builtin_amdgcn_s_wait_tensorcnt(0);
    __syncthreads();  // tile[cur] visible to all waves
    if (w == 0 && k0 + 32 < K) {
      unsigned nb = lds0 + (cur ^ 1) * 16384;
      tdm_load_2d(nb,        A + (size_t)m_blk * K + k0 + 32, 32, 128, (unsigned long long)K);
      tdm_load_2d(nb + 8192, W + (size_t)n_blk * K + k0 + 32, 32, 128, (unsigned long long)K);
    }
    const bf16* Ac = smem + cur * 8192;
    const bf16* Wc = Ac + 4096;

    v16bf b0 = load_frag_b(Wc, 32, nw + lr,      0, half);
    v16bf b1 = load_frag_b(Wc, 32, nw + 16 + lr, 0, half);
#pragma unroll
    for (int mi = 0; mi < 4; ++mi) {
      v16bf a = load_frag_a(Ac, 32, mw + mi * 16 + lr, 0, half);
      acc[mi][0] = wmma_bf16(a, b0, acc[mi][0]);
      acc[mi][1] = wmma_bf16(a, b1, acc[mi][1]);
    }
    __syncthreads();  // all waves done reading tile[cur]
    cur ^= 1;
  }

#pragma unroll
  for (int mi = 0; mi < 4; ++mi)
#pragma unroll
    for (int ni = 0; ni < 2; ++ni)
#pragma unroll
      for (int r = 0; r < 8; ++r) {
        int m = m_blk + mw + mi * 16 + 8 * half + r;
        int n = n_blk + nw + ni * 16 + lr;
        int64_t idx = (int64_t)(m >> 11) * Sb + (int64_t)(m & 2047) * St +
                      (int64_t)(n >> 6) * Sh + (int64_t)(n & 63) * Sd;
        store_out(out + idx, acc[mi][ni][r] * out_scale);
      }
}

// ---------------------------------------------------------------------
// Fallback GEMM (direct global fragments, fp32 converted on the fly)
// ---------------------------------------------------------------------
template <typename TA, typename TW, typename TO>
__global__ __launch_bounds__(256) void gemm_direct_kernel(
    const TA* __restrict__ A, const TW* __restrict__ W, TO* __restrict__ out,
    int K, float out_scale, int64_t Sb, int64_t St, int64_t Sh, int64_t Sd) {
  const int lane = threadIdx.x & 31;
  const int w    = threadIdx.x >> 5;
  const int half = lane >> 4;
  const int lr   = lane & 15;
  const int m_wave = blockIdx.y * 128 + (w & 1) * 64;
  const int n_wave = blockIdx.x * 128 + (w >> 1) * 32;

  v8f acc[4][2];
#pragma unroll
  for (int mi = 0; mi < 4; ++mi)
#pragma unroll
    for (int ni = 0; ni < 2; ++ni)
#pragma unroll
      for (int r = 0; r < 8; ++r) acc[mi][ni][r] = 0.0f;

  for (int k0 = 0; k0 < K; k0 += 32) {
    if (k0 + 32 < K) {
      __builtin_prefetch(A + (size_t)(m_wave + lr) * K + k0 + 32, 0, 1);
      __builtin_prefetch(W + (size_t)(n_wave + lr) * K + k0 + 32, 0, 1);
    }
    v16bf b0 = load_frag_b(W, K, n_wave + lr,      k0, half);
    v16bf b1 = load_frag_b(W, K, n_wave + 16 + lr, k0, half);
#pragma unroll
    for (int mi = 0; mi < 4; ++mi) {
      v16bf a = load_frag_a(A, K, m_wave + mi * 16 + lr, k0, half);
      acc[mi][0] = wmma_bf16(a, b0, acc[mi][0]);
      acc[mi][1] = wmma_bf16(a, b1, acc[mi][1]);
    }
  }
#pragma unroll
  for (int mi = 0; mi < 4; ++mi)
#pragma unroll
    for (int ni = 0; ni < 2; ++ni)
#pragma unroll
      for (int r = 0; r < 8; ++r) {
        int m = m_wave + mi * 16 + 8 * half + r;
        int n = n_wave + ni * 16 + lr;
        int64_t idx = (int64_t)(m >> 11) * Sb + (int64_t)(m & 2047) * St +
                      (int64_t)(n >> 6) * Sh + (int64_t)(n & 63) * Sd;
        store_out(out + idx, acc[mi][ni][r] * out_scale);
      }
}

// ---------------------------------------------------------------------
// Flash attention (causal). muP 1/d_head scale is pre-folded into Q.
// q,k: [B*T,C] bf16; vT: [B,H,D,T] bf16; y: [B*T,C] bf16.
// grid=(T/128, B*NHEAD), block=256; wave -> 16 query rows.
// Key loop split: mask-free bulk blocks + masked diagonal tail.
// ---------------------------------------------------------------------
__global__ __launch_bounds__(256) void attn_kernel(
    const bf16* __restrict__ q, const bf16* __restrict__ k,
    const bf16* __restrict__ vT, bf16* __restrict__ y) {
  const int lane = threadIdx.x & 31;
  const int w    = threadIdx.x >> 5;
  const int half = lane >> 4;
  const int lr   = lane & 15;
  const int h    = blockIdx.y & (NHEAD - 1);
  const int b    = blockIdx.y >> 4;
  const int q0   = blockIdx.x * 128 + w * 16;

  const bf16* qbase = q + (size_t)b * T_SEQ * C_DIM;
  const bf16* kbase = k + (size_t)b * T_SEQ * C_DIM;
  const bf16* vbase = vT + ((size_t)(b * NHEAD + h) * DHEAD) * T_SEQ;

  v16bf qf0 = load_frag_a(qbase, C_DIM, q0 + lr, h * DHEAD + 0,  half);
  v16bf qf1 = load_frag_a(qbase, C_DIM, q0 + lr, h * DHEAD + 32, half);

  v8f o[4];
  float mrow[8], lrow[8];
#pragma unroll
  for (int n0 = 0; n0 < 4; ++n0)
#pragma unroll
    for (int r = 0; r < 8; ++r) o[n0][r] = 0.0f;
#pragma unroll
  for (int r = 0; r < 8; ++r) { mrow[r] = -3.0e38f; lrow[r] = 0.0f; }

  __shared__ bf16 Pl[8][16][32];

  auto process_block = [&](int key0, bool do_mask) __attribute__((always_inline)) {
    // ---- S = Q K^T for two 16-key tiles ----
    v8f s[2];
#pragma unroll
    for (int j = 0; j < 2; ++j) {
      v16bf kf0 = load_frag_b(kbase, C_DIM, key0 + j * 16 + lr, h * DHEAD + 0,  half);
      v16bf kf1 = load_frag_b(kbase, C_DIM, key0 + j * 16 + lr, h * DHEAD + 32, half);
      v8f sj;
#pragma unroll
      for (int r = 0; r < 8; ++r) sj[r] = 0.0f;
      sj = wmma_bf16(qf0, kf0, sj);
      sj = wmma_bf16(qf1, kf1, sj);
      s[j] = sj;
    }

    float p0[8], p1[8], tmax[8], tsum[8], corr[8];
#pragma unroll
    for (int r = 0; r < 8; ++r) {
      float s0 = s[0][r];
      float s1 = s[1][r];
      if (do_mask) {  // compile-time folded after inlining
        int qrow = q0 + 8 * half + r;
        if (key0 + lr      > qrow) s0 = -3.0e38f;
        if (key0 + 16 + lr > qrow) s1 = -3.0e38f;
      }
      p0[r] = s0; p1[r] = s1;
      tmax[r] = fmaxf(s0, s1);
    }
#pragma unroll
    for (int off = 1; off < 16; off <<= 1)
#pragma unroll
      for (int r = 0; r < 8; ++r)
        tmax[r] = fmaxf(tmax[r], __shfl_xor(tmax[r], off, 32));
#pragma unroll
    for (int r = 0; r < 8; ++r) {
      float mnew = fmaxf(mrow[r], tmax[r]);
      corr[r] = __expf(mrow[r] - mnew);
      mrow[r] = mnew;
      p0[r] = __expf(p0[r] - mnew);
      p1[r] = __expf(p1[r] - mnew);
      tsum[r] = p0[r] + p1[r];
    }
#pragma unroll
    for (int off = 1; off < 16; off <<= 1)
#pragma unroll
      for (int r = 0; r < 8; ++r)
        tsum[r] += __shfl_xor(tsum[r], off, 32);
#pragma unroll
    for (int r = 0; r < 8; ++r) lrow[r] = lrow[r] * corr[r] + tsum[r];
#pragma unroll
    for (int n0 = 0; n0 < 4; ++n0)
#pragma unroll
      for (int r = 0; r < 8; ++r) o[n0][r] *= corr[r];

    // relayout P (C layout) -> A fragment via per-wave LDS
#pragma unroll
    for (int r = 0; r < 8; ++r) {
      Pl[w][8 * half + r][lr]      = (bf16)p0[r];
      Pl[w][8 * half + r][16 + lr] = (bf16)p1[r];
    }
    const bf16* prow = &Pl[w][lr][0];
    bf16x8 plo = *(const bf16x8*)(prow + 8 * half);
    bf16x8 phi = *(const bf16x8*)(prow + 16 + 8 * half);
    v16bf pf;
#pragma unroll
    for (int i = 0; i < 8; ++i) { pf[i] = plo[i]; pf[i + 8] = phi[i]; }

#pragma unroll
    for (int n0 = 0; n0 < 4; ++n0) {
      v16bf vf = load_frag_b(vbase, T_SEQ, n0 * 16 + lr, key0, half);
      o[n0] = wmma_bf16(pf, vf, o[n0]);
    }
  };

  const int nfull = (q0 + 1) >> 5;   // blocks with key0+31 <= q0: no masking
  const int nkb   = (q0 + 47) >> 5;  // all causal blocks
  for (int kb = 0; kb < nfull; ++kb) process_block(kb * 32, false);
  for (int kb = nfull; kb < nkb; ++kb) process_block(kb * 32, true);

  float inv_l[8];
#pragma unroll
  for (int r = 0; r < 8; ++r) inv_l[r] = 1.0f / lrow[r];
#pragma unroll
  for (int n0 = 0; n0 < 4; ++n0)
#pragma unroll
    for (int r = 0; r < 8; ++r) {
      size_t idx = (size_t)(b * T_SEQ + q0 + 8 * half + r) * C_DIM +
                   h * DHEAD + n0 * 16 + lr;
      y[idx] = (bf16)(o[n0][r] * inv_l[r]);
    }
}

// ---------------------------------------------------------------------
extern "C" void kernel_launch(void* const* d_in, const int* in_sizes, int n_in,
                              void* d_out, int out_size, void* d_ws, size_t ws_size,
                              hipStream_t stream) {
  const float* x     = (const float*)d_in[0];
  const float* wq    = (const float*)d_in[1];
  const float* wk    = (const float*)d_in[2];
  const float* wv    = (const float*)d_in[3];
  const float* wproj = (const float*)d_in[4];

  const int B = 2, M = B * T_SEQ;          // 4096
  const size_t MC = (size_t)M * C_DIM;     // 4M elems
  const size_t WC = (size_t)C_DIM * C_DIM; // 1M elems
  const float qscale = 1.0f / (float)DHEAD;  // muP 1/d_head folded into Q

  bf16* qb  = (bf16*)d_ws;
  bf16* kb  = qb + MC;
  bf16* vTb = kb + MC;
  bf16* yb  = vTb + MC;

  dim3 blk(256);
  dim3 gg(C_DIM / 128, M / 128);  // (8, 32)

  const int64_t Sb = (int64_t)T_SEQ * C_DIM, St = C_DIM, Sh = DHEAD, Sd = 1;
  const int64_t SbV = (int64_t)NHEAD * DHEAD * T_SEQ, StV = 1,
                ShV = (int64_t)DHEAD * T_SEQ, SdV = T_SEQ;

  const size_t need = (4 * MC + MC + 4 * WC) * sizeof(bf16);  // 48 MB
  if (ws_size >= need) {
    bf16* xb  = yb + MC;
    bf16* wqb = xb + MC;
    bf16* wkb = wqb + WC;
    bf16* wvb = wkb + WC;
    bf16* wpb = wvb + WC;
    cvt_bf16_kernel<<<(int)(MC / 8 / 256), blk, 0, stream>>>(x, xb, (int)(MC / 8));
    cvt_bf16_kernel<<<(int)(WC / 8 / 256), blk, 0, stream>>>(wq, wqb, (int)(WC / 8));
    cvt_bf16_kernel<<<(int)(WC / 8 / 256), blk, 0, stream>>>(wk, wkb, (int)(WC / 8));
    cvt_bf16_kernel<<<(int)(WC / 8 / 256), blk, 0, stream>>>(wv, wvb, (int)(WC / 8));
    cvt_bf16_kernel<<<(int)(WC / 8 / 256), blk, 0, stream>>>(wproj, wpb, (int)(WC / 8));

    gemm_tdm_kernel<bf16><<<gg, blk, 0, stream>>>(xb, wqb, qb,  C_DIM, qscale, Sb,  St,  Sh,  Sd);
    gemm_tdm_kernel<bf16><<<gg, blk, 0, stream>>>(xb, wkb, kb,  C_DIM, 1.0f,   Sb,  St,  Sh,  Sd);
    gemm_tdm_kernel<bf16><<<gg, blk, 0, stream>>>(xb, wvb, vTb, C_DIM, 1.0f,   SbV, StV, ShV, SdV);
    attn_kernel<<<dim3(T_SEQ / 128, B * NHEAD), blk, 0, stream>>>(qb, kb, vTb, yb);
    gemm_tdm_kernel<float><<<gg, blk, 0, stream>>>(yb, wpb, (float*)d_out,
                                                   C_DIM, 1.0f, Sb, St, Sh, Sd);
  } else {
    gemm_direct_kernel<float, float, bf16><<<gg, blk, 0, stream>>>(x, wq, qb,  C_DIM, qscale, Sb,  St,  Sh,  Sd);
    gemm_direct_kernel<float, float, bf16><<<gg, blk, 0, stream>>>(x, wk, kb,  C_DIM, 1.0f,   Sb,  St,  Sh,  Sd);
    gemm_direct_kernel<float, float, bf16><<<gg, blk, 0, stream>>>(x, wv, vTb, C_DIM, 1.0f,   SbV, StV, ShV, SdV);
    attn_kernel<<<dim3(T_SEQ / 128, B * NHEAD), blk, 0, stream>>>(qb, kb, vTb, yb);
    gemm_direct_kernel<bf16, float, float><<<gg, blk, 0, stream>>>(yb, wproj, (float*)d_out,
                                                                   C_DIM, 1.0f, Sb, St, Sh, Sd);
  }
}
